// SquaredExpModel_4544075399731
// MI455X (gfx1250) — matrix operand order repaired and checked by reading, not compiled, not run on hardware
//
#include <hip/hip_runtime.h>

typedef __attribute__((ext_vector_type(16))) _Float16 v16h;
typedef __attribute__((ext_vector_type(8)))  _Float16 v8h;
typedef __attribute__((ext_vector_type(8)))  float    v8f;
typedef __attribute__((ext_vector_type(4)))  float    v4f;

#define N_MODEL 10000
#define N_DATA  552
#define KPAD    10016   // N_MODEL padded to multiple of 32
#define NPAD    576     // N_DATA padded to multiple of 16
#define PROW    576     // P row stride in floats

// ---------------------------------------------------------------- utilities
__device__ __forceinline__ float waveReduceSum(float v) {
#pragma unroll
  for (int off = 16; off; off >>= 1) v += __shfl_xor(v, off, 32);
  return v;
}

__device__ __forceinline__ v16h combine16(v8h lo, v8h hi) {
  v16h r;
#pragma unroll
  for (int i = 0; i < 8; ++i) { r[i] = lo[i]; r[i + 8] = hi[i]; }
  return r;
}

// ------------------------------------------------- K0a: F (fp32) -> Ft (f16)
// Ft layout: [n (576)][k (10016)], zero padded. Rows of Ft are rows of F.
__global__ __launch_bounds__(256)
void k_conv_F(const float* __restrict__ F, _Float16* __restrict__ Ft) {
  size_t idx = (size_t)blockIdx.x * 256 + threadIdx.x; // 576*10016 total
  int n = (int)(idx / KPAD);
  int k = (int)(idx % KPAD);
  float v = (n < N_DATA && k < N_MODEL) ? F[(size_t)n * N_MODEL + k] : 0.0f;
  Ft[idx] = (_Float16)v;
}

// --------------------- K1: fused  P = (sigma^2 * exp(-c*dist)) @ F^T  (WMMA)
// 157 blocks x 256 threads (8 waves). Block computes 64 rows x 576 cols.
// Per K-step the WG stages A (64x32 exp'd f16) and B (32x576 of F^T) in LDS;
// all WMMA operands come from LDS. Next step's staging data is prefetched
// into registers during the WMMA block. Waves: 2 M-groups x 4 N-groups,
// each wave = 2 M-subtiles x 9 N-tiles = 18 accumulators.
__global__ __launch_bounds__(256)
void k_fused_exp_gemm(const float* __restrict__ dist,
                      const _Float16* __restrict__ Ft,
                      float* __restrict__ P,
                      const float* __restrict__ p_ls,
                      const float* __restrict__ p_sigma) {
  __shared__ __align__(16) _Float16 ldsA[64 * 32];    //  4 KB
  __shared__ __align__(16) _Float16 ldsB[576 * 32];   // 36 KB

  const int tid  = threadIdx.x;
  const int lane = tid & 31;
  const int w    = tid >> 5;      // 0..7
  const int mg   = w >> 2;        // 0..1  M-group (rows mg*32..+31)
  const int ng   = w & 3;         // 0..3  N-group (cols ng*144..+143)
  const int m0g  = blockIdx.x * 64;

  const float ls   = *p_ls;
  const float sg   = *p_sigma;
  const float negc = -0.5f / (ls * ls);
  const float s2   = sg * sg;

  v8f acc[2][9];
  {
    v8f z = {};
#pragma unroll
    for (int s = 0; s < 2; ++s)
#pragma unroll
      for (int f = 0; f < 9; ++f) acc[s][f] = z;
  }

  // A staging role: thread -> dist row (tid>>2), 8-wide k chunk ((tid&3)*8)
  const int sr = tid >> 2;
  const int sc = (tid & 3) * 8;
  long grow = m0g + sr;
  if (grow > N_MODEL - 1) grow = N_MODEL - 1;   // clamp (stores are guarded)
  const float* drow = dist + grow * (long)N_MODEL;
  // B staging role: chunks c = i*256 + tid (i<9); row = c>>2, off = (c&3)*8

  const int fsel = (lane < 16) ? 0 : 8;
  const int bsel = (lane < 16) ? 0 : 16;

  // prologue: register-load staging data for K-step 0
  v4f da0 = *(const v4f*)(drow + sc);
  v4f da1 = *(const v4f*)(drow + sc + 4);
  v8h bst[9];
#pragma unroll
  for (int i = 0; i < 9; ++i) {
    int c = i * 256 + tid;
    bst[i] = *(const v8h*)(Ft + (size_t)(c >> 2) * KPAD + (c & 3) * 8);
  }

  for (int kk = 0; kk < 313; ++kk) {
    const int k0 = kk * 32;

    __syncthreads();   // previous step's LDS reads complete (WAR)

    // commit staged registers -> LDS (exp applied to A here)
    {
      const bool valid = (k0 + sc) < N_MODEL;  // 10000%8==0: chunk all-or-none
      v8h hv;
#pragma unroll
      for (int j = 0; j < 8; ++j) {
        float dj = (j < 4) ? da0[j] : da1[j - 4];
        hv[j] = valid ? (_Float16)(s2 * __expf(negc * dj)) : (_Float16)0.0f;
      }
      *(v8h*)&ldsA[sr * 32 + sc] = hv;
#pragma unroll
      for (int i = 0; i < 9; ++i) {
        int c = i * 256 + tid;
        *(v8h*)&ldsB[(c >> 2) * 32 + (c & 3) * 8] = bst[i];
      }
    }
    __syncthreads();   // staged tile visible (RAW)

    // issue next step's global loads now; latency hides under the WMMA block
    if (kk < 312) {
      const int kn = k0 + 32;
      if (kn + sc < N_MODEL) {
        da0 = *(const v4f*)(drow + kn + sc);
        da1 = *(const v4f*)(drow + kn + sc + 4);
      }
#pragma unroll
      for (int i = 0; i < 9; ++i) {
        int c = i * 256 + tid;
        bst[i] = *(const v8h*)(Ft + (size_t)(c >> 2) * KPAD + kn + (c & 3) * 8);
      }
      int kp = k0 + 256 + sc;                   // ~8 steps ahead into L2
      if (kp < N_MODEL) __builtin_prefetch(drow + kp, 0, 3);
    }

    // A fragments for this wave's two M-subtiles (ISA 16-bit A layout)
    const int r0 = (mg * 2 + 0) * 16 + (lane & 15);
    const int r1 = (mg * 2 + 1) * 16 + (lane & 15);
    v16h a0 = combine16(*(const v8h*)&ldsA[r0 * 32 + fsel],
                        *(const v8h*)&ldsA[r0 * 32 + 16 + fsel]);
    v16h a1 = combine16(*(const v8h*)&ldsA[r1 * 32 + fsel],
                        *(const v8h*)&ldsA[r1 * 32 + 16 + fsel]);

#pragma unroll
    for (int f = 0; f < 9; ++f) {
      const int n = ng * 144 + f * 16 + (lane & 15);
      v16h b = combine16(*(const v8h*)&ldsB[n * 32 + fsel],
                         *(const v8h*)&ldsB[n * 32 + 16 + fsel]);
      acc[0][f] = __builtin_amdgcn_wmma_f32_16x16x32_f16(
          false, a0, false, b, (short)0, acc[0][f], false, false);
      acc[1][f] = __builtin_amdgcn_wmma_f32_16x16x32_f16(
          false, a1, false, b, (short)0, acc[1][f], false, false);
    }
  }

  // store P  (C layout: N = lane%16, M = vgpr + 8*(lane>=16))
  const int col0 = ng * 144 + (lane & 15);
#pragma unroll
  for (int s = 0; s < 2; ++s) {
    const int rtop = m0g + (mg * 2 + s) * 16 + ((lane < 16) ? 0 : 8);
#pragma unroll
    for (int f = 0; f < 9; ++f) {
#pragma unroll
      for (int v = 0; v < 8; ++v) {
        int r = rtop + v;
        if (r < N_MODEL) P[(size_t)r * PROW + col0 + f * 16] = acc[s][f][v];
      }
    }
  }
}

// ----------------------------------- K0b: P (fp32, [k][n]) -> Pt (f16, [n][k])
__global__ __launch_bounds__(256)
void k_conv_P(const float* __restrict__ P, _Float16* __restrict__ Pt) {
  size_t idx = (size_t)blockIdx.x * 256 + threadIdx.x; // 576*10016 total
  int n = (int)(idx / KPAD);
  int k = (int)(idx % KPAD);
  float v = (k < N_MODEL) ? P[(size_t)k * PROW + n] : 0.0f;
  Pt[idx] = (_Float16)v;
}

// -------------------------- K2: A2 = data_cov + F @ P (WMMA, identity-padded)
// 324 waves, each owns 1 row-tile x 4 col-tiles (A fragment reused 4x).
__global__ __launch_bounds__(256)
void k_inner_gemm(const _Float16* __restrict__ Ft,
                  const _Float16* __restrict__ Pt,
                  const float* __restrict__ data_cov,
                  float* __restrict__ A2) {
  const int lane = threadIdx.x & 31;
  const int wid  = blockIdx.x * 8 + (threadIdx.x >> 5);  // 0..327
  if (wid >= 324) return;                                 // no barriers used
  const int ti  = wid / 9;        // 0..35
  const int tjg = wid % 9;        // col tiles tjg*4 .. +3
  const int fsel = (lane < 16) ? 0 : 8;
  const int bsel = (lane < 16) ? 0 : 16;

  const _Float16* ar = Ft + (size_t)(ti * 16 + (lane & 15)) * KPAD;
  const _Float16* br[4];
#pragma unroll
  for (int t = 0; t < 4; ++t)
    br[t] = Pt + (size_t)((tjg * 4 + t) * 16 + (lane & 15)) * KPAD;

  v8f acc[4];
  {
    v8f z = {};
#pragma unroll
    for (int t = 0; t < 4; ++t) acc[t] = z;
  }

  for (int kk = 0; kk < 313; ++kk) {
    const int k0 = kk * 32;
    v8h al = *(const v8h*)(ar + k0 + fsel);
    v8h ah = *(const v8h*)(ar + k0 + 16 + fsel);
    v8h bl[4], bh[4];
#pragma unroll
    for (int t = 0; t < 4; ++t) {
      bl[t] = *(const v8h*)(br[t] + k0 + fsel);
      bh[t] = *(const v8h*)(br[t] + k0 + 16 + fsel);
    }
    v16h a = combine16(al, ah);
#pragma unroll
    for (int t = 0; t < 4; ++t)
      acc[t] = __builtin_amdgcn_wmma_f32_16x16x32_f16(
          false, a, false, combine16(bl[t], bh[t]), (short)0, acc[t],
          false, false);
  }

  const int i0 = ti * 16 + ((lane < 16) ? 0 : 8);
#pragma unroll
  for (int t = 0; t < 4; ++t) {
    const int j = (tjg * 4 + t) * 16 + (lane & 15);
#pragma unroll
    for (int v = 0; v < 8; ++v) {
      int i = i0 + v;
      float add;
      if (i < N_DATA && j < N_DATA) add = data_cov[(size_t)i * N_DATA + j];
      else                          add = (i == j) ? 1.0f : 0.0f; // inert pad
      A2[(size_t)i * NPAD + j] = acc[t][v] + add;
    }
  }
}

// --------------------------- K3: pm[i] = d_obs[i] - m0 * rowsum(F[i]) (pad 0)
__global__ __launch_bounds__(256)
void k_misfit(const float* __restrict__ F, const float* __restrict__ d_obs,
              const float* __restrict__ p_m0, float* __restrict__ pm) {
  int lane = threadIdx.x & 31;
  int row  = blockIdx.x * 8 + (threadIdx.x >> 5); // 0..575
  float s = 0.0f;
  if (row < N_DATA) {
    const float* fr = F + (size_t)row * N_MODEL;
    for (int k = lane; k < N_MODEL; k += 32) s += fr[k];
  }
  s = waveReduceSum(s);
  if (lane == 0) pm[row] = (row < N_DATA) ? (d_obs[row] - (*p_m0) * s) : 0.0f;
}

// ------------- K4: in-place LU (no pivot), logdet, tri-solves, log-likelihood
__global__ __launch_bounds__(1024)
void k_lu_solve(float* __restrict__ A, const float* __restrict__ pm,
                float* __restrict__ tmpv, float* __restrict__ out) {
  __shared__ float xs[NPAD];
  volatile float* x = xs;
  const int lane = threadIdx.x & 31;
  const int w    = threadIdx.x >> 5;   // 0..31

  for (int j = 0; j < NPAD - 1; ++j) {
    __syncthreads();                    // step j-1 fully committed
    float inv = 1.0f / A[(size_t)j * NPAD + j];
    const float* prow = A + (size_t)j * NPAD;
    for (int i = j + 1 + w; i < NPAD; i += 32) {
      float* irow = A + (size_t)i * NPAD;
      float l = irow[j] * inv;          // broadcast load completes before store
      for (int k = j + 1 + lane; k < NPAD; k += 32)
        irow[k] -= l * prow[k];
      if (lane == 0) irow[j] = l;
    }
  }
  __syncthreads();

  if (w == 0) {                         // single wave: in-order LDS, no barriers
    float s = 0.0f;
    for (int j = lane; j < N_DATA; j += 32)
      s += __logf(fabsf(A[(size_t)j * NPAD + j]));
    float logdet = waveReduceSum(s);

    // forward solve L y = pm (unit diagonal)
    for (int i = 0; i < NPAD; ++i) {
      float a = 0.0f;
      for (int jj = lane; jj < i; jj += 32)
        a += A[(size_t)i * NPAD + jj] * x[jj];
      a = waveReduceSum(a);
      if (lane == 0) x[i] = pm[i] - a;
    }
    // back solve U t = y
    for (int i = NPAD - 1; i >= 0; --i) {
      float a = 0.0f;
      for (int jj = i + 1 + lane; jj < NPAD; jj += 32)
        a += A[(size_t)i * NPAD + jj] * x[jj];
      a = waveReduceSum(a);
      if (lane == 0) x[i] = (x[i] - a) / A[(size_t)i * NPAD + i];
    }
    for (int i = lane; i < NPAD; i += 32) tmpv[i] = x[i];
    float d = 0.0f;
    for (int i = lane; i < NPAD; i += 32) d += pm[i] * x[i];
    d = waveReduceSum(d);
    if (lane == 0) out[0] = logdet + d;  // log-likelihood
  }
}

// ------------------------------ K5: m_posterior = m0 + P @ tmp (wave per row)
__global__ __launch_bounds__(256)
void k_posterior(const float* __restrict__ P, const float* __restrict__ tmpv,
                 const float* __restrict__ p_m0, float* __restrict__ out) {
  int lane = threadIdx.x & 31;
  int row  = blockIdx.x * 8 + (threadIdx.x >> 5);
  if (row >= N_MODEL) return;
  const float* pr = P + (size_t)row * PROW;
  float s = 0.0f;
  for (int n = lane; n < PROW; n += 32) s += pr[n] * tmpv[n]; // pads are 0
  s = waveReduceSum(s);
  if (lane == 0) out[1 + row] = *p_m0 + s;
}

// ---------------------------------------------------------------------------
extern "C" void kernel_launch(void* const* d_in, const int* in_sizes, int n_in,
                              void* d_out, int out_size, void* d_ws, size_t ws_size,
                              hipStream_t stream) {
  const float* dist     = (const float*)d_in[0];  // [10000,10000]
  const float* F        = (const float*)d_in[1];  // [552,10000]
  const float* d_obs    = (const float*)d_in[2];  // [552]
  const float* data_cov = (const float*)d_in[3];  // [552,552]
  const float* m0       = (const float*)d_in[4];  // scalar
  const float* ls       = (const float*)d_in[5];  // scalar
  const float* sigma    = (const float*)d_in[6];  // scalar

  // workspace carve-up (all offsets 256B aligned); total ~47.5 MB
  char* ws = (char*)d_ws;
  _Float16* Ft  = (_Float16*)(ws);                 // 576*10016*2 = 11,538,432
  float*    P   = (float*)  (ws + 11538432);       // 10000*576*4 = 23,040,000
  _Float16* Pt  = (_Float16*)(ws + 34578432);      // 11,538,432
  float*    A2  = (float*)  (ws + 46116864);       // 576*576*4   =  1,327,104
  float*    pm  = (float*)  (ws + 47443968);       // 576*4
  float*    tmpv= (float*)  (ws + 47446272);       // 576*4
  float*    out = (float*)d_out;                   // [0]=loglik, [1..10000]=m_post

  k_conv_F        <<<22536, 256, 0, stream>>>(F, Ft);
  k_fused_exp_gemm<<<157,   256, 0, stream>>>(dist, Ft, P, ls, sigma);
  k_conv_P        <<<22536, 256, 0, stream>>>(P, Pt);
  k_inner_gemm    <<<41,    256, 0, stream>>>(Ft, Pt, data_cov, A2);
  k_misfit        <<<72,    256, 0, stream>>>(F, d_obs, m0, pm);
  k_lu_solve      <<<1,    1024, 0, stream>>>(A2, pm, tmpv, out);
  k_posterior     <<<1250,  256, 0, stream>>>(P, tmpv, m0, out);
}
